// MQParallelFFTransformer_16604343567000
// MI455X (gfx1250) — matrix operand order repaired and checked by reading, not compiled, not run on hardware
//
#include <hip/hip_runtime.h>

typedef unsigned short u16;
typedef __attribute__((ext_vector_type(16))) __bf16 v16bf;
typedef __attribute__((ext_vector_type(2)))  __bf16 v2bf;
typedef __attribute__((ext_vector_type(8)))  float  v8f;

union Frag16 { v16bf v; uint4 u[2]; };

static __device__ __forceinline__ u16 f2bf(float f) {
  union { float f; unsigned u; } c; c.f = f;
  unsigned u = c.u;
  unsigned r = u + 0x7FFFu + ((u >> 16) & 1u);
  return (u16)(r >> 16);
}
static __device__ __forceinline__ float bf2f(u16 h) {
  union { unsigned u; float f; } c; c.u = ((unsigned)h) << 16;
  return c.f;
}
// Packed f32->bf16 pair: low 16 = a, high 16 = b.
static __device__ __forceinline__ unsigned pk_bf16(float a, float b) {
#if __has_builtin(__builtin_amdgcn_cvt_pk_bf16_f32)
  union { v2bf v; unsigned u; } c;
  c.v = __builtin_amdgcn_cvt_pk_bf16_f32(a, b);
  return c.u;
#else
  return (unsigned)f2bf(a) | ((unsigned)f2bf(b) << 16);
#endif
}

// ---- CDNA5 async global->LDS copies (ASYNCcnt-tracked) --------------------
static __device__ __forceinline__ void async_ld32(unsigned lds, unsigned voff,
                                                  unsigned long long base) {
  asm volatile(
      "global_load_async_to_lds_b128 %0, %1, %2\n\t"
      "global_load_async_to_lds_b128 %0, %1, %2 offset:16"
      :: "v"(lds), "v"(voff), "s"(base) : "memory");
}
static __device__ __forceinline__ void async_ld16(unsigned lds, unsigned voff,
                                                  unsigned long long base) {
  asm volatile("global_load_async_to_lds_b128 %0, %1, %2"
               :: "v"(lds), "v"(voff), "s"(base) : "memory");
}
static __device__ __forceinline__ void wait_async0() {
  asm volatile("s_wait_asynccnt 0" ::: "memory");
}
static __device__ __forceinline__ unsigned lds_off(const void* p) {
  return (unsigned)(uintptr_t)p;  // LDS aperture: logical addr = addr[31:0]
}

// ---- DPP16 reductions within each 16-lane half ----------------------------
template <int CTRL>
static __device__ __forceinline__ float dpp_mov(float x) {
  return __int_as_float(
      __builtin_amdgcn_update_dpp(0, __float_as_int(x), CTRL, 0xF, 0xF, true));
}
static __device__ __forceinline__ float red16_max(float x) {
  x = fmaxf(x, dpp_mov<0xB1>(x));   // quad_perm [1,0,3,2]
  x = fmaxf(x, dpp_mov<0x4E>(x));   // quad_perm [2,3,0,1]
  x = fmaxf(x, dpp_mov<0x141>(x));  // row_half_mirror
  x = fmaxf(x, dpp_mov<0x140>(x));  // row_mirror
  return x;
}
static __device__ __forceinline__ float red16_sum(float x) {
  x += dpp_mov<0xB1>(x);
  x += dpp_mov<0x4E>(x);
  x += dpp_mov<0x141>(x);
  x += dpp_mov<0x140>(x);
  return x;
}

// A-fragment (16x32 bf16): lane holds row = lane%16.
static __device__ __forceinline__ v16bf load_frag_a(const u16* p, int ld, int lane) {
  int r = lane & 15, hi = (lane >> 4) & 1;
  const u16* b = p + (size_t)r * ld + hi * 8;
  Frag16 t;
  t.u[0] = *(const uint4*)(b);
  t.u[1] = *(const uint4*)(b + 16);
  return t.v;
}
// B-fragment (32x16 bf16) from Bt (N x K row-major): lane holds col = lane%16.
static __device__ __forceinline__ v16bf load_frag_b(const u16* p, int ld, int lane) {
  int r = lane & 15, hi = (lane >> 4) & 1;
  const u16* b = p + (size_t)r * ld + hi * 8;
  Frag16 t;
  t.u[0] = *(const uint4*)(b);
  t.u[1] = *(const uint4*)(b + 16);
  return t.v;
}

static __device__ __forceinline__ v8f wmma_bf16(v16bf a, v16bf b, v8f c) {
  return __builtin_amdgcn_wmma_f32_16x16x32_bf16(false, a, false, b, (short)0, c,
                                                 false, false);
}

// ---------------------------------------------------------------------------
// Weight convert + transpose: Wt[n*K + k] = bf16(W[k*N + n])
// ---------------------------------------------------------------------------
__global__ __launch_bounds__(256) void cvtT_kernel(const float* __restrict__ W,
                                                   u16* __restrict__ Wt,
                                                   int K, int N) {
  size_t i = (size_t)blockIdx.x * 256 + threadIdx.x;
  int k = (int)(i / (size_t)N);
  int n = (int)(i % (size_t)N);
  Wt[(size_t)n * K + k] = f2bf(W[i]);
}

// ---------------------------------------------------------------------------
// LayerNorm over last dim (1024). One block per row, 256 threads x 4 elems.
// ---------------------------------------------------------------------------
__global__ __launch_bounds__(256) void ln_kernel(const float* __restrict__ in,
                                                 const float* __restrict__ gamma,
                                                 u16* __restrict__ obf,
                                                 float* __restrict__ of32) {
  __shared__ float ss[8], sq[8];
  int row = blockIdx.x, tid = threadIdx.x;
  const float4 v = ((const float4*)(in + (size_t)row * 1024))[tid];
  float s = v.x + v.y + v.z + v.w;
  float q = v.x * v.x + v.y * v.y + v.z * v.z + v.w * v.w;
  s = red16_sum(s); s += __shfl_xor(s, 16);
  q = red16_sum(q); q += __shfl_xor(q, 16);
  if ((tid & 31) == 0) { ss[tid >> 5] = s; sq[tid >> 5] = q; }
  __syncthreads();
  s = 0.f; q = 0.f;
#pragma unroll
  for (int i = 0; i < 8; ++i) { s += ss[i]; q += sq[i]; }
  float mu = s * (1.f / 1024.f);
  float var = q * (1.f / 1024.f) - mu * mu;
  float rs = rsqrtf(var + 1e-5f);
  const float4 g4 = ((const float4*)gamma)[tid];
  float o0 = (v.x - mu) * rs * g4.x;
  float o1 = (v.y - mu) * rs * g4.y;
  float o2 = (v.z - mu) * rs * g4.z;
  float o3 = (v.w - mu) * rs * g4.w;
  if (obf) {
    uint2 pk;
    pk.x = pk_bf16(o0, o1);
    pk.y = pk_bf16(o2, o3);
    *(uint2*)(obf + (size_t)row * 1024 + tid * 4) = pk;
  }
  if (of32) {
    ((float4*)(of32 + (size_t)row * 1024))[tid] = make_float4(o0, o1, o2, o3);
  }
}

// ---------------------------------------------------------------------------
// Generic WMMA GEMM: C(MxN) = A(MxK bf16 row-major) @ Bt(NxK bf16 row-major)^T
// 8 waves (2x4). Wave tile 64xWN (WN=64 for N>=256, 32 for the KV GEMM).
// Block tile 128 x 4*WN, K-step 32, double-buffered async global->LDS.
// ---------------------------------------------------------------------------
#define LDP 40  // padded LDS row stride: 16 consecutive rows hit distinct banks

template <int EPI, int WN>
__global__ __launch_bounds__(256) void gemm_kernel(const u16* __restrict__ A,
                                                   const u16* __restrict__ Bt,
                                                   void* __restrict__ o1,
                                                   void* __restrict__ o2,
                                                   const void* __restrict__ aux,
                                                   int M, int N, int K,
                                                   float scale) {
  constexpr int BN = 4 * WN;       // block tile N
  constexpr int NJ = WN / 16;      // B frags per wave
  __shared__ u16 lA[2][128 * LDP];
  __shared__ u16 lB[2][BN * LDP];
  const int tid = threadIdx.x, lane = tid & 31, w = tid >> 5;
  const int wm = w >> 2, wn = w & 3;  // 2 x 4 wave grid
  const int row0 = blockIdx.y * 128;
  const int col0 = blockIdx.x * BN;

  v8f c[4][NJ];
#pragma unroll
  for (int i = 0; i < 4; ++i)
#pragma unroll
    for (int j = 0; j < NJ; ++j) c[i][j] = (v8f)(0.f);

  const int lrow = tid >> 1;
  const int lhalf = (tid & 1) * 16;
  const unsigned long long baseA = (unsigned long long)(uintptr_t)A;
  const unsigned long long baseB = (unsigned long long)(uintptr_t)Bt;

  auto issue = [&](int buf, int k0) {
    unsigned la = lds_off(&lA[buf][lrow * LDP + lhalf]);
    unsigned va = (unsigned)(((size_t)(row0 + lrow) * K + k0 + lhalf) * 2);
    async_ld32(la, va, baseA);
    if constexpr (WN == 64) {
      // B tile 256 rows x 32: one row (64 B) per thread
      unsigned lb = lds_off(&lB[buf][tid * LDP]);
      unsigned vb = (unsigned)(((size_t)(col0 + tid) * K + k0) * 2);
      async_ld32(lb, vb, baseB);
      async_ld32(lb + 32, vb + 32, baseB);
    } else {
      unsigned lb = lds_off(&lB[buf][lrow * LDP + lhalf]);
      unsigned vb = (unsigned)(((size_t)(col0 + lrow) * K + k0 + lhalf) * 2);
      async_ld32(lb, vb, baseB);
    }
  };

  issue(0, 0);
  int cur = 0;
  for (int k0 = 0; k0 < K; k0 += 32) {
    wait_async0();
    __syncthreads();
    if (k0 + 32 < K) issue(cur ^ 1, k0 + 32);

    v16bf af[4], bfr[NJ];
#pragma unroll
    for (int i = 0; i < 4; ++i)
      af[i] = load_frag_a(&lA[cur][(wm * 64 + 16 * i) * LDP], LDP, lane);
#pragma unroll
    for (int j = 0; j < NJ; ++j)
      bfr[j] = load_frag_b(&lB[cur][(wn * WN + 16 * j) * LDP], LDP, lane);
#pragma unroll
    for (int i = 0; i < 4; ++i)
#pragma unroll
      for (int j = 0; j < NJ; ++j) c[i][j] = wmma_bf16(af[i], bfr[j], c[i][j]);
    cur ^= 1;
  }

  const int r = lane & 15, hi = (lane >> 4) & 1;
#pragma unroll
  for (int i = 0; i < 4; ++i) {
#pragma unroll
    for (int j = 0; j < NJ; ++j) {
      const int rbase = row0 + wm * 64 + 16 * i + 8 * hi;
      const int gcol = col0 + wn * WN + 16 * j + r;
      if constexpr (EPI == 0 || EPI == 3) {
#pragma unroll
        for (int g = 0; g < 8; g += 2) {
          float v0 = c[i][j][g] * scale;
          float v1 = c[i][j][g + 1] * scale;
          if constexpr (EPI == 3) {
            float g0 = bf2f(((const u16*)aux)[(size_t)(rbase + g) * N + gcol]);
            float g1 =
                bf2f(((const u16*)aux)[(size_t)(rbase + g + 1) * N + gcol]);
            v0 *= g0 / (1.f + __expf(-g0));
            v1 *= g1 / (1.f + __expf(-g1));
          }
          unsigned pk = pk_bf16(v0, v1);
          ((u16*)o1)[(size_t)(rbase + g) * N + gcol] = (u16)pk;
          ((u16*)o1)[(size_t)(rbase + g + 1) * N + gcol] = (u16)(pk >> 16);
        }
      } else {
#pragma unroll
        for (int g = 0; g < 8; ++g) {
          int grow = rbase + g;
          float val = c[i][j][g] * scale;
          if constexpr (EPI == 1) {
            if (gcol < 64) {
              ((u16*)o1)[(size_t)grow * 64 + gcol] = f2bf(val);
            } else {
              int b = grow >> 10, jj = grow & 1023, d = gcol - 64;
              ((u16*)o2)[((size_t)b * 64 + d) * 1024 + jj] = f2bf(val);
            }
          } else if constexpr (EPI == 2) {
            ((float*)o1)[(size_t)grow * N + gcol] =
                val + ((const float*)aux)[(size_t)grow * N + gcol];
          } else if constexpr (EPI == 4) {
            float* po = (float*)o1 + (size_t)grow * N + gcol;
            *po = *po + val;
          }
        }
      }
    }
  }
}

// ---------------------------------------------------------------------------
// Multi-query flash attention. q: (4096,1024) bf16 (pre-scaled).
// k: (B*1024,64) bf16. vT: per-batch (64,1024) bf16. out: (4096,1024) bf16.
// Grid (8, 16 heads, 4 batches); 8 waves x 16 query rows.
// ---------------------------------------------------------------------------
#define KLD 72  // k tile row stride (32 keys x 64 dims)
#define VLD 40  // vT tile row stride (64 dims x 32 keys)

__global__ __launch_bounds__(256) void attn_kernel(const u16* __restrict__ q,
                                                   const u16* __restrict__ k,
                                                   const u16* __restrict__ vT,
                                                   u16* __restrict__ out) {
  __shared__ u16 lk[2][32 * KLD];
  __shared__ u16 lv[2][64 * VLD];
  __shared__ u16 lp[8][16 * 32];  // per-wave P tile (16x32)
  const int tid = threadIdx.x, lane = tid & 31, w = tid >> 5;
  const int r = lane & 15, hi = (lane >> 4) & 1;
  const int h = blockIdx.y, b = blockIdx.z;
  const int qrow0 = blockIdx.x * 128 + w * 16;

  const u16* qp = q + ((size_t)(b * 1024 + qrow0)) * 1024 + h * 64;
  const v16bf aq0 = load_frag_a(qp, 1024, lane);
  const v16bf aq1 = load_frag_a(qp + 32, 1024, lane);

  const unsigned long long kbase =
      (unsigned long long)(uintptr_t)(k + (size_t)b * 1024 * 64);
  const unsigned long long vbase =
      (unsigned long long)(uintptr_t)(vT + (size_t)b * 64 * 1024);

  const int kkey = tid >> 3, koct = tid & 7;
  const int vd = tid >> 2, vquad = tid & 3;
  auto issue = [&](int buf, int j0) {
    unsigned lka = lds_off(&lk[buf][kkey * KLD + koct * 8]);
    unsigned vka = (unsigned)((((size_t)(j0 + kkey)) * 64 + koct * 8) * 2);
    async_ld16(lka, vka, kbase);
    unsigned lva = lds_off(&lv[buf][vd * VLD + vquad * 8]);
    unsigned vva = (unsigned)((((size_t)vd) * 1024 + j0 + vquad * 8) * 2);
    async_ld16(lva, vva, vbase);
  };

  v8f acc[4];
#pragma unroll
  for (int t = 0; t < 4; ++t) acc[t] = (v8f)(0.f);
  float m[8], l[8];
#pragma unroll
  for (int g = 0; g < 8; ++g) { m[g] = -3.0e38f; l[g] = 0.f; }

  u16* lpw = &lp[w][0];
  issue(0, 0);
  int cur = 0;
  for (int j0 = 0; j0 < 1024; j0 += 32) {
    wait_async0();
    __syncthreads();
    if (j0 + 32 < 1024) issue(cur ^ 1, j0 + 32);

    v8f s0 = (v8f)(0.f), s1 = (v8f)(0.f);
    {
      v16bf k0a = load_frag_b(&lk[cur][0], KLD, lane);
      v16bf k0b = load_frag_b(&lk[cur][32], KLD, lane);
      s0 = wmma_bf16(aq0, k0a, s0);
      s0 = wmma_bf16(aq1, k0b, s0);
      v16bf k1a = load_frag_b(&lk[cur][16 * KLD], KLD, lane);
      v16bf k1b = load_frag_b(&lk[cur][16 * KLD + 32], KLD, lane);
      s1 = wmma_bf16(aq0, k1a, s1);
      s1 = wmma_bf16(aq1, k1b, s1);
    }
#pragma unroll
    for (int g = 0; g < 8; ++g) {
      float mx = red16_max(fmaxf(s0[g], s1[g]));
      float mn = fmaxf(m[g], mx);
      float alpha = __expf(m[g] - mn);
      float p0 = __expf(s0[g] - mn);
      float p1 = __expf(s1[g] - mn);
      float sum = red16_sum(p0 + p1);
      l[g] = l[g] * alpha + sum;
      m[g] = mn;
#pragma unroll
      for (int t = 0; t < 4; ++t) acc[t][g] = acc[t][g] * alpha;
      int Mr = g + 8 * hi;
      unsigned pk = pk_bf16(p0, p1);
      lpw[Mr * 32 + r] = (u16)pk;
      lpw[Mr * 32 + 16 + r] = (u16)(pk >> 16);
    }
    // wave-local LDS round trip converts C-layout P into an A-fragment
    asm volatile("s_wait_dscnt 0" ::: "memory");
    v16bf pa = load_frag_a(lpw, 32, lane);
#pragma unroll
    for (int t = 0; t < 4; ++t) {
      v16bf vf = load_frag_b(&lv[cur][t * 16 * VLD], VLD, lane);
      acc[t] = wmma_bf16(pa, vf, acc[t]);
    }
    cur ^= 1;
  }

  float rl[8];
#pragma unroll
  for (int g = 0; g < 8; ++g) rl[g] = 1.f / l[g];
#pragma unroll
  for (int t = 0; t < 4; ++t) {
#pragma unroll
    for (int g = 0; g < 8; g += 2) {
      unsigned pk = pk_bf16(acc[t][g] * rl[g], acc[t][g + 1] * rl[g + 1]);
      size_t row = (size_t)(b * 1024 + qrow0 + g + 8 * hi);
      size_t col = (size_t)h * 64 + t * 16 + r;
      out[row * 1024 + col] = (u16)pk;
      out[(row + 1) * 1024 + col] = (u16)(pk >> 16);
    }
  }
}

// ---------------------------------------------------------------------------
// Host launcher
// ---------------------------------------------------------------------------
extern "C" void kernel_launch(void* const* d_in, const int* in_sizes, int n_in,
                              void* d_out, int out_size, void* d_ws,
                              size_t ws_size, hipStream_t stream) {
  const float* x    = (const float*)d_in[0];
  const float* y    = (const float*)d_in[1];
  const float* g1   = (const float*)d_in[2];
  const float* g2   = (const float*)d_in[3];
  const float* gout = (const float*)d_in[4];
  const float* Wq   = (const float*)d_in[5];
  const float* Wkv  = (const float*)d_in[6];
  const float* Wo   = (const float*)d_in[7];
  const float* Wff1 = (const float*)d_in[8];
  const float* Wff2 = (const float*)d_in[9];
  float* outp = (float*)d_out;

  char* base = (char*)d_ws;
  size_t off = 0;
  auto alloc = [&](size_t bytes) -> char* {
    char* p = base + off;
    off += (bytes + 255) & ~(size_t)255;
    return p;
  };
  u16* wqT   = (u16*)alloc((size_t)1024 * 1024 * 2);
  u16* wkvT  = (u16*)alloc((size_t)128 * 1024 * 2);
  u16* woT   = (u16*)alloc((size_t)1024 * 1024 * 2);
  u16* wff1T = (u16*)alloc((size_t)8192 * 1024 * 2);
  u16* wff2T = (u16*)alloc((size_t)1024 * 4096 * 2);
  u16* xn    = (u16*)alloc((size_t)4096 * 1024 * 2);
  u16* ynb   = (u16*)alloc((size_t)4096 * 1024 * 2);
  float* ynf = (float*)alloc((size_t)4096 * 1024 * 4);
  u16* qb    = (u16*)alloc((size_t)4096 * 1024 * 2);
  u16* kb    = (u16*)alloc((size_t)4096 * 64 * 2);
  u16* vT    = (u16*)alloc((size_t)4 * 64 * 1024 * 2);
  u16* ao    = (u16*)alloc((size_t)4096 * 1024 * 2);
  float* pre = (float*)alloc((size_t)4096 * 1024 * 4);
  u16* gate  = (u16*)alloc((size_t)4096 * 4096 * 2);
  u16* act   = (u16*)alloc((size_t)4096 * 4096 * 2);

  cvtT_kernel<<<(1024 * 1024) / 256, 256, 0, stream>>>(Wq, wqT, 1024, 1024);
  cvtT_kernel<<<(1024 * 128) / 256, 256, 0, stream>>>(Wkv, wkvT, 1024, 128);
  cvtT_kernel<<<(1024 * 1024) / 256, 256, 0, stream>>>(Wo, woT, 1024, 1024);
  cvtT_kernel<<<(1024 * 8192) / 256, 256, 0, stream>>>(Wff1, wff1T, 1024, 8192);
  cvtT_kernel<<<(4096 * 1024) / 256, 256, 0, stream>>>(Wff2, wff2T, 4096, 1024);

  ln_kernel<<<4096, 256, 0, stream>>>(x, g1, xn, nullptr);
  ln_kernel<<<4096, 256, 0, stream>>>(y, g2, ynb, ynf);

  gemm_kernel<0, 64><<<dim3(4, 32), 256, 0, stream>>>(
      xn, wqT, qb, nullptr, nullptr, 4096, 1024, 1024, 0.125f);
  gemm_kernel<1, 32><<<dim3(1, 32), 256, 0, stream>>>(
      ynb, wkvT, kb, vT, nullptr, 4096, 128, 1024, 1.0f);

  attn_kernel<<<dim3(8, 16, 4), 256, 0, stream>>>(qb, kb, vT, ao);

  gemm_kernel<2, 64><<<dim3(4, 32), 256, 0, stream>>>(
      ao, woT, pre, nullptr, ynf, 4096, 1024, 1024, 1.0f);
  ln_kernel<<<4096, 256, 0, stream>>>(pre, gout, nullptr, outp);

  gemm_kernel<0, 64><<<dim3(16, 32), 256, 0, stream>>>(
      ynb, wff1T + (size_t)4096 * 1024, gate, nullptr, nullptr, 4096, 4096,
      1024, 1.0f);
  gemm_kernel<3, 64><<<dim3(16, 32), 256, 0, stream>>>(
      ynb, wff1T, act, nullptr, gate, 4096, 4096, 1024, 1.0f);
  gemm_kernel<4, 64><<<dim3(4, 32), 256, 0, stream>>>(
      act, wff2T, d_out, nullptr, nullptr, 4096, 1024, 4096, 1.0f);
}
